// PointNetFeaturePropagation1_81295140979492
// MI455X (gfx1250) — matrix operand order repaired and checked by reading, not compiled, not run on hardware
//
#include <hip/hip_runtime.h>
#include <stdint.h>

// ---------------------------------------------------------------------------
// PointNet Feature Propagation for MI455X (gfx1250, wave32, WMMA + TDM).
// GEMMs: bf16 WMMA fed by Tensor Data Mover (B tiles) and async global->LDS
// copies (A tiles), double buffered. Pipeline is HBM-bound (~1GB @ 23.3TB/s),
// so all other kernels are coalesced/vectorized b128 passes.
// ---------------------------------------------------------------------------

#define B_    8
#define N_    16384
#define S_    1024
#define D1_   256
#define D2_   256
#define CIN_  512
#define CMID_ 256
#define COUT_ 256
#define NCOL  (B_ * N_)          // 131072 columns
#define BN_EPS   1e-5f
#define DIST_EPS 1e-8f

typedef __attribute__((ext_vector_type(16))) __bf16        bf16x16;
typedef __attribute__((ext_vector_type(8)))  float         f32x8;
typedef __attribute__((ext_vector_type(4)))  unsigned int  u32x4;
typedef __attribute__((ext_vector_type(8)))  unsigned int  u32x8;

union FragU { u32x4 q[2]; bf16x16 v; };

__device__ __forceinline__ unsigned short f2bf(float f) {
  unsigned int u = __float_as_uint(f);
  u += 0x7FFFu + ((u >> 16) & 1u);               // round-to-nearest-even
  return (unsigned short)(u >> 16);
}
__device__ __forceinline__ float bf2f(unsigned short h) {
  return __uint_as_float(((unsigned int)h) << 16);
}

// ---------------- weight conversion f32 -> bf16 ----------------
__global__ void k_cvt_w(const float* __restrict__ w0, const float* __restrict__ w1,
                        unsigned short* __restrict__ dst) {
  int i = blockIdx.x * 256 + threadIdx.x;
  const int n0 = CMID_ * CIN_;                   // 131072
  const int nt = n0 + CMID_ * CMID_;             // 196608
  if (i < nt) {
    float v = (i < n0) ? w0[i] : w1[i - n0];
    dst[i] = f2bf(v);
  }
}

// ---------------- points2 [B,D2,S] -> p2t [B,S,D2] (f32, tiled transpose) ----
__global__ __launch_bounds__(256) void k_transpose_p2(const float* __restrict__ p2,
                                                      float* __restrict__ p2t) {
  __shared__ float t[32][33];
  int blk = blockIdx.x;
  int sb = blk & 31;                 // S/32 = 32
  int cb = (blk >> 5) & 7;           // D2/32 = 8
  int b  = blk >> 8;
  int s0 = sb * 32, c0 = cb * 32;
  int tx = threadIdx.x & 31, ty = threadIdx.x >> 5;
#pragma unroll
  for (int r = 0; r < 4; ++r) {
    int c = ty + 8 * r;
    t[c][tx] = p2[(b * D2_ + c0 + c) * S_ + s0 + tx];
  }
  __syncthreads();
#pragma unroll
  for (int r = 0; r < 4; ++r) {
    int s = ty + 8 * r;
    p2t[(b * S_ + s0 + s) * D2_ + c0 + tx] = t[tx][s];
  }
}

// ---------------- 3-NN search: one thread per query point ----------------
__global__ __launch_bounds__(256) void k_knn(const float* __restrict__ xyz1,
                                             const float* __restrict__ xyz2,
                                             int* __restrict__ idx,
                                             float* __restrict__ wgt) {
  __shared__ float sx[S_], sy[S_], sz[S_];
  int b = blockIdx.x >> 6;                       // N/256 = 64 blocks per batch
  int n = ((blockIdx.x & 63) << 8) + threadIdx.x;
  const float* x2 = xyz2 + b * 3 * S_;
  for (int i = threadIdx.x; i < S_; i += 256) {
    sx[i] = x2[i];
    sy[i] = x2[S_ + i];
    sz[i] = x2[2 * S_ + i];
  }
  __syncthreads();
  float px = xyz1[(b * 3 + 0) * N_ + n];
  float py = xyz1[(b * 3 + 1) * N_ + n];
  float pz = xyz1[(b * 3 + 2) * N_ + n];
  float d0 = 3.4e38f, d1 = 3.4e38f, d2 = 3.4e38f;
  int   i0 = 0, i1 = 0, i2 = 0;
  for (int s = 0; s < S_; ++s) {                 // broadcast LDS reads
    float dx = px - sx[s], dy = py - sy[s], dz = pz - sz[s];
    float d = dx * dx + dy * dy + dz * dz;
    if (d < d0)      { d2 = d1; i2 = i1; d1 = d0; i1 = i0; d0 = d; i0 = s; }
    else if (d < d1) { d2 = d1; i2 = i1; d1 = d;  i1 = s; }
    else if (d < d2) { d2 = d;  i2 = s; }
  }
  float r0 = 1.f / (d0 + DIST_EPS);
  float r1 = 1.f / (d1 + DIST_EPS);
  float r2 = 1.f / (d2 + DIST_EPS);
  float inv = 1.f / (r0 + r1 + r2);
  int c3 = (b * N_ + n) * 3;
  idx[c3] = i0; idx[c3 + 1] = i1; idx[c3 + 2] = i2;
  wgt[c3] = r0 * inv; wgt[c3 + 1] = r1 * inv; wgt[c3 + 2] = r2 * inv;
}

// ---------------- points1 [B,D1,N] -> X1[col*512 + c] (bf16, transpose) -----
__global__ __launch_bounds__(256) void k_part1(const float* __restrict__ p1,
                                               unsigned short* __restrict__ X1) {
  __shared__ float t[32][33];
  int blk = blockIdx.x;
  int nb = blk & 511;                // N/32 = 512
  int cb = (blk >> 9) & 7;           // D1/32 = 8
  int b  = blk >> 12;
  int n0 = nb * 32, c0 = cb * 32;
  int tx = threadIdx.x & 31, ty = threadIdx.x >> 5;
#pragma unroll
  for (int r = 0; r < 4; ++r) {
    int c = ty + 8 * r;
    t[c][tx] = p1[(b * D1_ + c0 + c) * N_ + n0 + tx];
  }
  __syncthreads();
#pragma unroll
  for (int r = 0; r < 4; ++r) {
    int n = ty + 8 * r;
    X1[(size_t)(b * N_ + n0 + n) * CIN_ + c0 + tx] = f2bf(t[tx][n]);
  }
}

// ---------------- weighted gather: X1[col*512 + 256 + c] ----------------
__global__ __launch_bounds__(256) void k_interp(const float* __restrict__ p2t,
                                                const int* __restrict__ idx,
                                                const float* __restrict__ wgt,
                                                unsigned short* __restrict__ X1) {
  int c = threadIdx.x;               // 0..255 feature channel
  int base = blockIdx.x * 8;
#pragma unroll 2
  for (int p = 0; p < 8; ++p) {
    int col = base + p;
    int b = col >> 14;               // col / N_
    int c3 = col * 3;
    int j0 = idx[c3], j1 = idx[c3 + 1], j2 = idx[c3 + 2];
    float w0 = wgt[c3], w1 = wgt[c3 + 1], w2 = wgt[c3 + 2];
    const float* r0 = p2t + (size_t)(b * S_ + j0) * D2_;
    const float* r1 = p2t + (size_t)(b * S_ + j1) * D2_;
    const float* r2 = p2t + (size_t)(b * S_ + j2) * D2_;
    float v = w0 * r0[c] + w1 * r1[c] + w2 * r2[c];
    X1[(size_t)col * CIN_ + D1_ + c] = f2bf(v);
  }
}

// ---------------- bf16 WMMA GEMM: Y[col*256+m] = W[256xK] * X[col*K+k] ------
// Block tile 128(M) x 128(cols), 8 waves in a 2x4 grid, each wave 64x32
// (4x2 fragments of v_wmma_f32_16x16x32_bf16).
//   B tiles (X, streaming): TDM tensor_load_to_lds, 2D tile 32x128, with
//     pad_enable (interval=16 DW, amount=4 DW) producing the stride-40
//     conflict-free LDS layout directly. Tracked by TENSORcnt.
//   A tiles (W): per-lane global_load_async_to_lds_b128, tracked by ASYNCcnt.
// Both double buffered: issue tile i+1, wait down to tile i, barrier, compute.
template <int K>
__global__ __launch_bounds__(256) void k_gemm(const unsigned short* __restrict__ W,
                                              const unsigned short* __restrict__ X,
                                              unsigned short* __restrict__ Y) {
  __shared__ __align__(16) unsigned short As[2][128 * 40];
  __shared__ __align__(16) unsigned short Bs[2][128 * 40];
  __shared__ __align__(16) unsigned short Cs[128 * 136];

  const int tid  = threadIdx.x;
  const int lane = tid & 31;
  const int wave = tid >> 5;
  const int wm = wave >> 2;          // 0..1 -> M offset 64*wm
  const int wn = wave & 3;           // 0..3 -> col offset 32*wn
  const int half = lane >> 4;
  const int lm = lane & 15;
  const int m0   = blockIdx.y * 128;
  const int col0 = blockIdx.x * 128;

  // A tile: 512 chunks of 8 bf16; thread handles chunks tid and tid+256.
  auto issueA = [&](int k0, int buf) {
#pragma unroll
    for (int it = 0; it < 2; ++it) {
      int cc = tid + it * 256;
      int row = cc >> 2, seg = cc & 3;
      const unsigned short* g = &W[(size_t)(m0 + row) * K + k0 + seg * 8];
      unsigned int l = (unsigned int)(size_t)&As[buf][row * 40 + seg * 8];
      asm volatile("global_load_async_to_lds_b128 %0, %1, off"
                   :: "v"(l), "v"(g) : "memory");
    }
  };

  // B tile via Tensor Data Mover: one descriptor, issued by wave 0 only.
  auto issueB = [&](int k0, int buf) {
    if (wave == 0) {
      unsigned long long ga = (unsigned long long)(size_t)&X[(size_t)col0 * K + k0];
      unsigned int lds = (unsigned int)(size_t)&Bs[buf][0];
      u32x4 g0;
      g0[0] = 1u;                                            // count=1, user mode
      g0[1] = lds;                                           // lds_addr
      g0[2] = (unsigned int)ga;                              // global_addr[31:0]
      g0[3] = ((unsigned int)(ga >> 32) & 0x01FFFFFFu) | 0x80000000u; // [56:32] | type=2
      u32x8 g1;
      g1[0] = (1u << 16)       // data_size = 2 bytes
            | (1u << 20)       // pad_enable
            | (3u << 22)       // pad_interval: 16 DWORDs (= 32 bf16 row)
            | (3u << 25);      // pad_amount: 4 DWORDs (= 8 bf16 -> stride 40)
      g1[1] = ((unsigned)K & 0xFFFFu) << 16;                 // tensor_dim0 lo16
      g1[2] = (((unsigned)K >> 16) & 0xFFFFu)                // tensor_dim0 hi16
            | (((unsigned)NCOL & 0xFFFFu) << 16);            // tensor_dim1 lo16
      g1[3] = (((unsigned)NCOL >> 16) & 0xFFFFu)             // tensor_dim1 hi16
            | (32u << 16);                                   // tile_dim0 = 32
      g1[4] = 128u;                                          // tile_dim1=128, tile_dim2=0
      g1[5] = (unsigned)K;                                   // tensor_dim0_stride lo32
      g1[6] = 0u;                                            // stride hi / dim1_stride lo
      g1[7] = 0u;
      asm volatile("tensor_load_to_lds %0, %1" :: "s"(g0), "s"(g1) : "memory");
    }
  };

  constexpr int NT = K / 32;
  issueA(0, 0);
  issueB(0, 0);

  f32x8 acc[4][2] = {};

  for (int i = 0; i < NT; ++i) {
    const int buf = i & 1;
    if (i + 1 < NT) {
      issueA((i + 1) * 32, buf ^ 1);
      issueB((i + 1) * 32, buf ^ 1);
      // 2 async chunks + 1 tensor op in flight for tile i+1; wait for tile i.
      asm volatile("s_wait_asynccnt 0x2" ::: "memory");
      asm volatile("s_wait_tensorcnt 0x1" ::: "memory");  // no-op on waves w/ cnt 0
    } else {
      asm volatile("s_wait_asynccnt 0x0" ::: "memory");
      asm volatile("s_wait_tensorcnt 0x0" ::: "memory");
    }
    __syncthreads();

    FragU bfr[2];
#pragma unroll
    for (int fn = 0; fn < 2; ++fn) {
      const unsigned short* p = &Bs[buf][(wn * 32 + fn * 16 + lm) * 40 + half * 8];
      bfr[fn].q[0] = *(const u32x4*)p;
      bfr[fn].q[1] = *(const u32x4*)(p + 16);
    }
#pragma unroll
    for (int fm = 0; fm < 4; ++fm) {
      FragU afr;
      const unsigned short* p = &As[buf][(wm * 64 + fm * 16 + lm) * 40 + half * 8];
      afr.q[0] = *(const u32x4*)p;
      afr.q[1] = *(const u32x4*)(p + 16);
#pragma unroll
      for (int fn = 0; fn < 2; ++fn) {
        acc[fm][fn] = __builtin_amdgcn_wmma_f32_16x16x32_bf16(
            false, afr.v, false, bfr[fn].v, (short)0, acc[fm][fn], false, false);
      }
    }
    __syncthreads();
  }

  // stage C (transposed to [col][m]) in LDS, then coalesced bf16 stores
#pragma unroll
  for (int fm = 0; fm < 4; ++fm) {
#pragma unroll
    for (int fn = 0; fn < 2; ++fn) {
      int mb = wm * 64 + fm * 16 + half * 8;
      int nl = wn * 32 + fn * 16 + lm;
#pragma unroll
      for (int r = 0; r < 8; ++r)
        Cs[nl * 136 + mb + r] = f2bf(acc[fm][fn][r]);
    }
  }
  __syncthreads();
#pragma unroll
  for (int it = 0; it < 8; ++it) {
    int cc = tid + it * 256;         // 2048 chunks of 8 bf16
    int n = cc >> 4, seg = cc & 15;
    *(u32x4*)&Y[(size_t)(col0 + n) * CMID_ + m0 + seg * 8] =
        *(const u32x4*)&Cs[n * 136 + seg * 8];
  }
}

// ---------------- per-channel sum / sumsq partials (deterministic) ----------
__global__ __launch_bounds__(256) void k_stats(const unsigned short* __restrict__ Y,
                                               float* __restrict__ part) {
  int c = threadIdx.x;
  size_t col0 = (size_t)blockIdx.x * 512;
  float s = 0.f, s2 = 0.f;
  for (int i = 0; i < 512; ++i) {
    float v = bf2f(Y[(col0 + i) * CMID_ + c]);
    s += v; s2 += v * v;
  }
  part[blockIdx.x * 512 + c]       = s;
  part[blockIdx.x * 512 + 256 + c] = s2;
}

// ---------------- reduce partials -> per-channel affine (scale, shift) ------
__global__ void k_affine(const float* __restrict__ part, const float* __restrict__ g,
                         const float* __restrict__ be, float2* __restrict__ aff) {
  int c = threadIdx.x;
  float s = 0.f, s2 = 0.f;
  for (int k = 0; k < 256; ++k) {    // 256 blocks of partials, fixed order
    s  += part[k * 512 + c];
    s2 += part[k * 512 + 256 + c];
  }
  float mean = s * (1.f / (float)NCOL);
  float var  = s2 * (1.f / (float)NCOL) - mean * mean;
  float inv  = rsqrtf(var + BN_EPS);
  float sc = g[c] * inv;
  aff[c] = make_float2(sc, be[c] - mean * sc);  // conv bias cancels under BN
}

// ---------------- in-place BN + ReLU on bf16 Y (vec8) ----------------
__global__ __launch_bounds__(256) void k_bn_relu(unsigned short* __restrict__ Y,
                                                 const float2* __restrict__ aff) {
  __shared__ float2 sa[256];
  sa[threadIdx.x] = aff[threadIdx.x];
  __syncthreads();
  size_t i = (size_t)blockIdx.x * 256 + threadIdx.x;  // vec8 index
  size_t base = i * 8;
  int c0 = (int)(base & 255);
  u32x4 q = *(const u32x4*)(Y + base);
  u32x4 o;
#pragma unroll
  for (int j = 0; j < 4; ++j) {
    unsigned int u = q[j];
    float2 a0 = sa[c0 + 2 * j], a1 = sa[c0 + 2 * j + 1];
    float lo = fmaxf(bf2f((unsigned short)(u & 0xFFFFu)) * a0.x + a0.y, 0.f);
    float hi = fmaxf(bf2f((unsigned short)(u >> 16)) * a1.x + a1.y, 0.f);
    o[j] = (unsigned int)f2bf(lo) | ((unsigned int)f2bf(hi) << 16);
  }
  *(u32x4*)(Y + base) = o;
}

// ---------------- final BN + ReLU + transpose to out [B,256,N] f32 ----------
__global__ __launch_bounds__(256) void k_final(const unsigned short* __restrict__ Y2,
                                               const float2* __restrict__ aff,
                                               float* __restrict__ out) {
  __shared__ float t[32][33];
  int blk = blockIdx.x;
  int nb = blk & 511;                // N/32
  int cb = (blk >> 9) & 7;           // 256/32
  int b  = blk >> 12;
  int n0 = nb * 32, c0 = cb * 32;
  int tx = threadIdx.x & 31, ty = threadIdx.x >> 5;
  float2 a = aff[c0 + tx];
#pragma unroll
  for (int r = 0; r < 4; ++r) {
    int nl = ty + 8 * r;
    size_t col = (size_t)b * N_ + n0 + nl;
    float v = bf2f(Y2[col * COUT_ + c0 + tx]);
    t[nl][tx] = fmaxf(v * a.x + a.y, 0.f);
  }
  __syncthreads();
#pragma unroll
  for (int r = 0; r < 4; ++r) {
    int cl = ty + 8 * r;
    out[(size_t)(b * COUT_ + c0 + cl) * N_ + n0 + tx] = t[tx][cl];
  }
}

// ---------------------------------------------------------------------------
extern "C" void kernel_launch(void* const* d_in, const int* in_sizes, int n_in,
                              void* d_out, int out_size, void* d_ws, size_t ws_size,
                              hipStream_t stream) {
  (void)in_sizes; (void)n_in; (void)out_size; (void)ws_size;
  const float* xyz1    = (const float*)d_in[0];
  const float* xyz2    = (const float*)d_in[1];
  const float* points1 = (const float*)d_in[2];
  const float* points2 = (const float*)d_in[3];
  const float* w0      = (const float*)d_in[4];
  const float* g0      = (const float*)d_in[6];
  const float* be0     = (const float*)d_in[7];
  const float* w1      = (const float*)d_in[8];
  const float* g1      = (const float*)d_in[10];
  const float* be1     = (const float*)d_in[11];
  float* out = (float*)d_out;

  // workspace layout (all offsets 16B-aligned); Y2 aliases dead X1 region
  char* ws = (char*)d_ws;
  float2*         aff0  = (float2*)(ws + 0);                     //   2 KB
  float2*         aff1  = (float2*)(ws + 2048);                  //   2 KB
  unsigned short* W0bf  = (unsigned short*)(ws + 4096);          // 256 KB
  unsigned short* W1bf  = W0bf + CMID_ * CIN_;                   // 128 KB
  int*            idx   = (int*)(ws + 397312);                   // 1.5 MB
  float*          wgt   = (float*)(ws + 1970176);                // 1.5 MB
  float*          p2t   = (float*)(ws + 3543040);                //   8 MB
  float*          part  = (float*)(ws + 11931648);               // 512 KB
  unsigned short* X1    = (unsigned short*)(ws + 12455936);      // 128 MB
  unsigned short* Y1    = (unsigned short*)(ws + 146673664);     //  64 MB
  unsigned short* Y2    = X1;  // X1 dead after GEMM1 -> reuse   (total ~204 MB)

  k_cvt_w       <<<768,   256, 0, stream>>>(w0, w1, W0bf);
  k_transpose_p2<<<2048,  256, 0, stream>>>(points2, p2t);
  k_knn         <<<512,   256, 0, stream>>>(xyz1, xyz2, idx, wgt);
  k_part1       <<<32768, 256, 0, stream>>>(points1, X1);
  k_interp      <<<16384, 256, 0, stream>>>(p2t, idx, wgt, X1);

  k_gemm<CIN_>  <<<dim3(NCOL / 128, 2), 256, 0, stream>>>(W0bf, X1, Y1);
  k_stats       <<<256,   256, 0, stream>>>(Y1, part);
  k_affine      <<<1,     256, 0, stream>>>(part, g0, be0, aff0);
  k_bn_relu     <<<16384, 256, 0, stream>>>(Y1, aff0);

  k_gemm<CMID_> <<<dim3(NCOL / 128, 2), 256, 0, stream>>>(W1bf, Y1, Y2);
  k_stats       <<<256,   256, 0, stream>>>(Y2, part);
  k_affine      <<<1,     256, 0, stream>>>(part, g1, be1, aff1);
  k_final       <<<32768, 256, 0, stream>>>(Y2, aff1, out);
}
